// GPTLanguageModel_68547678044782
// MI455X (gfx1250) — matrix operand order repaired and checked by reading, not compiled, not run on hardware
//
#include <hip/hip_runtime.h>
#include <hip/hip_bf16.h>

// Problem constants (from the reference)
#define V_    32000
#define E_    1024
#define H_    16
#define L_    4
#define T_    1024
#define B_    4
#define HS_   64
#define DFF_  4096
#define NTOK  (B_ * T_)   // 4096

typedef __attribute__((ext_vector_type(16))) __bf16 v16bf;
typedef __attribute__((ext_vector_type(8)))  float  v8f;

// ---------------------------------------------------------------------------
// WMMA helpers (gfx1250: D = A(16x32 bf16) * B(32x16 bf16) + C(16x16 f32))
// ---------------------------------------------------------------------------
__device__ __forceinline__ v8f wmma_bf16(v16bf a, v16bf b, v8f c) {
  return __builtin_amdgcn_wmma_f32_16x16x32_bf16(
      /*neg_a=*/false, a, /*neg_b=*/false, b,
      /*c_mod=*/(short)0, c, /*reuse_a=*/false, /*reuse_b=*/false);
}

// A-matrix 16x32 bf16 lane layout (matching B):
//   lane group g = lane>>4;  element e -> K = 16g + e
// -> two contiguous 8-element (16B) runs at offsets 16g and 16g+8.
__device__ __forceinline__ v16bf ldfragA(const __bf16* row, int g) {
  union { v16bf v; uint4 u[2]; } r;
  r.u[0] = *reinterpret_cast<const uint4*>(row + 16 * g);
  r.u[1] = *reinterpret_cast<const uint4*>(row + 16 * g + 8);
  return r.v;
}

// B-matrix 32x16 bf16 lane layout (this code's k order, matching A; not AMD's table order):
//   lane n = lane&15 holds column n; K = 16g + e  (one contiguous 16-run).
__device__ __forceinline__ v16bf ldfragB(const __bf16* row, int g) {
  union { v16bf v; uint4 u[2]; } r;
  r.u[0] = *reinterpret_cast<const uint4*>(row + 16 * g);
  r.u[1] = *reinterpret_cast<const uint4*>(row + 16 * g + 8);
  return r.v;
}

// Same fragment patterns gathered from global fp32 with on-the-fly bf16 cvt.
// Loads batched into distinct registers first so they can all be in flight.
__device__ __forceinline__ v16bf gfragA(const float* row, int g) {
  float t[16];
#pragma unroll
  for (int i = 0; i < 8; ++i) t[i] = row[16 * g + i];
#pragma unroll
  for (int i = 0; i < 8; ++i) t[8 + i] = row[16 * g + 8 + i];
  v16bf f;
#pragma unroll
  for (int i = 0; i < 16; ++i) f[i] = (__bf16)t[i];
  return f;
}
__device__ __forceinline__ v16bf gfragB(const float* row, int g) {
  float t[16];
#pragma unroll
  for (int i = 0; i < 16; ++i) t[i] = row[16 * g + i];
  v16bf f;
#pragma unroll
  for (int i = 0; i < 16; ++i) f[i] = (__bf16)t[i];
  return f;
}

// ---------------------------------------------------------------------------
// GEMM: C[M,N] = A[M,K] @ W[K,N] (+bias[N]) (+relu) (+resid[M,N])
// 128x128 block tile, BK=32, 256 threads = 8 waves, wave tile 32x64.
// Double-buffered LDS; next tile's global loads issued before current WMMAs.
// ---------------------------------------------------------------------------
__global__ __launch_bounds__(256)
void gemm_bf16w(const float* __restrict__ A, const float* __restrict__ W,
                const float* __restrict__ bias, const float* __restrict__ resid,
                float* __restrict__ C, int M, int N, int K, int relu)
{
  __shared__ __bf16 As[2][128][32];
  __shared__ __bf16 Bs[2][128][32];   // stored [n][k] (transposed W tile)

  const int tid  = threadIdx.x;
  const int lane = tid & 31;
  const int w    = tid >> 5;
  const int g    = lane >> 4;
  const int ln   = lane & 15;
  const int m0   = blockIdx.y * 128;
  const int n0   = blockIdx.x * 128;
  const int mB   = (w & 3) * 32;   // wave row offset (4 waves down)
  const int nB   = (w >> 2) * 64;  // wave col offset (2 waves across)

  // staging thread mapping
  const int ar = tid >> 3;          // A row 0..31 (+p*32)
  const int ac = (tid & 7) * 4;     // A col 0..28
  const int bk = tid >> 5;          // W k-row 0..7 (+p*8)
  const int bc = (tid & 31) * 4;    // W n-col 0..124

  v8f acc[2][4] = {};
  const int ksteps = K >> 5;

  float4 ra[4], rb[4];
  // ---- prologue: stage tile 0 into buffer 0 ----
#pragma unroll
  for (int p = 0; p < 4; ++p)
    ra[p] = *reinterpret_cast<const float4*>(A + (size_t)(m0 + ar + p * 32) * K + ac);
#pragma unroll
  for (int p = 0; p < 4; ++p)
    rb[p] = *reinterpret_cast<const float4*>(W + (size_t)(bk + p * 8) * N + n0 + bc);
#pragma unroll
  for (int p = 0; p < 4; ++p) {
    As[0][ar + p * 32][ac + 0] = (__bf16)ra[p].x;
    As[0][ar + p * 32][ac + 1] = (__bf16)ra[p].y;
    As[0][ar + p * 32][ac + 2] = (__bf16)ra[p].z;
    As[0][ar + p * 32][ac + 3] = (__bf16)ra[p].w;
  }
#pragma unroll
  for (int p = 0; p < 4; ++p) {
    Bs[0][bc + 0][bk + p * 8] = (__bf16)rb[p].x;
    Bs[0][bc + 1][bk + p * 8] = (__bf16)rb[p].y;
    Bs[0][bc + 2][bk + p * 8] = (__bf16)rb[p].z;
    Bs[0][bc + 3][bk + p * 8] = (__bf16)rb[p].w;
  }

  for (int kt = 0; kt < ksteps; ++kt) {
    __syncthreads();
    const int  cur  = kt & 1;
    const bool more = (kt + 1) < ksteps;

    // ---- issue next tile's global loads (latency hides under WMMA) ----
    if (more) {
      const int ko = (kt + 1) * 32;
#pragma unroll
      for (int p = 0; p < 4; ++p)
        ra[p] = *reinterpret_cast<const float4*>(A + (size_t)(m0 + ar + p * 32) * K + ko + ac);
#pragma unroll
      for (int p = 0; p < 4; ++p)
        rb[p] = *reinterpret_cast<const float4*>(W + (size_t)(ko + bk + p * 8) * N + n0 + bc);
      if (kt + 2 < ksteps) {  // global_prefetch_b8 two tiles ahead
        __builtin_prefetch(A + (size_t)(m0 + ar) * K + (kt + 2) * 32 + ac, 0, 3);
        __builtin_prefetch(W + (size_t)((kt + 2) * 32 + bk) * N + n0 + bc, 0, 3);
      }
    }

    // ---- compute current tile ----
    v16bf af[2], bw[4];
#pragma unroll
    for (int i = 0; i < 2; ++i) af[i] = ldfragA(&As[cur][mB + i * 16 + ln][0], g);
#pragma unroll
    for (int j = 0; j < 4; ++j) bw[j] = ldfragB(&Bs[cur][nB + j * 16 + ln][0], g);
#pragma unroll
    for (int i = 0; i < 2; ++i)
#pragma unroll
      for (int j = 0; j < 4; ++j)
        acc[i][j] = wmma_bf16(af[i], bw[j], acc[i][j]);

    // ---- convert + store next tile into the other buffer ----
    if (more) {
      const int nx = cur ^ 1;
#pragma unroll
      for (int p = 0; p < 4; ++p) {
        As[nx][ar + p * 32][ac + 0] = (__bf16)ra[p].x;
        As[nx][ar + p * 32][ac + 1] = (__bf16)ra[p].y;
        As[nx][ar + p * 32][ac + 2] = (__bf16)ra[p].z;
        As[nx][ar + p * 32][ac + 3] = (__bf16)ra[p].w;
      }
#pragma unroll
      for (int p = 0; p < 4; ++p) {
        Bs[nx][bc + 0][bk + p * 8] = (__bf16)rb[p].x;
        Bs[nx][bc + 1][bk + p * 8] = (__bf16)rb[p].y;
        Bs[nx][bc + 2][bk + p * 8] = (__bf16)rb[p].z;
        Bs[nx][bc + 3][bk + p * 8] = (__bf16)rb[p].w;
      }
    }
  }

  // ---- epilogue: C/D layout: elem r -> row r+8g, col = lane&15 ----
#pragma unroll
  for (int i = 0; i < 2; ++i) {
#pragma unroll
    for (int j = 0; j < 4; ++j) {
#pragma unroll
      for (int r = 0; r < 8; ++r) {
        const int gr = m0 + mB + i * 16 + r + 8 * g;
        const int gc = n0 + nB + j * 16 + ln;
        float o = acc[i][j][r];
        if (bias)  o += bias[gc];
        if (relu)  o = fmaxf(o, 0.0f);
        if (resid) o += resid[(size_t)gr * N + gc];
        C[(size_t)gr * N + gc] = o;
      }
    }
  }
}

// ---------------------------------------------------------------------------
// Embedding: x[row] = tok_emb[index[row]] + pos_emb[row % T]
// ---------------------------------------------------------------------------
__global__ __launch_bounds__(256)
void embed_k(const int* __restrict__ idx, const float* __restrict__ tok,
             const float* __restrict__ pos, float* __restrict__ x)
{
  const int row = blockIdx.x;
  const int t   = row % T_;
  const int id  = idx[row];
  const int c   = threadIdx.x * 4;
  const float4 a = *reinterpret_cast<const float4*>(tok + (size_t)id * E_ + c);
  const float4 p = *reinterpret_cast<const float4*>(pos + (size_t)t * E_ + c);
  float4 o; o.x = a.x + p.x; o.y = a.y + p.y; o.z = a.z + p.z; o.w = a.w + p.w;
  *reinterpret_cast<float4*>(x + (size_t)row * E_ + c) = o;
}

// ---------------------------------------------------------------------------
// LayerNorm over E=1024 per row; 256 threads, 4 elems/thread, two-pass.
// ---------------------------------------------------------------------------
__global__ __launch_bounds__(256)
void layernorm_k(const float* __restrict__ x, const float* __restrict__ gw,
                 const float* __restrict__ gb, float* __restrict__ out)
{
  __shared__ float red[16];
  const int row  = blockIdx.x;
  const int tid  = threadIdx.x;
  const int lane = tid & 31;
  const int w    = tid >> 5;
  const float4 v = *reinterpret_cast<const float4*>(x + (size_t)row * E_ + tid * 4);

  float s = v.x + v.y + v.z + v.w;
#pragma unroll
  for (int o = 16; o > 0; o >>= 1) s += __shfl_xor(s, o, 32);
  if (lane == 0) red[w] = s;
  __syncthreads();
  float tot = 0.0f;
#pragma unroll
  for (int i = 0; i < 8; ++i) tot += red[i];
  const float mu = tot * (1.0f / E_);

  const float c0 = v.x - mu, c1 = v.y - mu, c2 = v.z - mu, c3 = v.w - mu;
  float s2 = c0 * c0 + c1 * c1 + c2 * c2 + c3 * c3;
#pragma unroll
  for (int o = 16; o > 0; o >>= 1) s2 += __shfl_xor(s2, o, 32);
  if (lane == 0) red[8 + w] = s2;
  __syncthreads();
  float var = 0.0f;
#pragma unroll
  for (int i = 0; i < 8; ++i) var += red[8 + i];
  var *= (1.0f / E_);
  const float rs = rsqrtf(var + 1e-5f);

  const int c = tid * 4;
  float4 o;
  o.x = c0 * rs * gw[c + 0] + gb[c + 0];
  o.y = c1 * rs * gw[c + 1] + gb[c + 1];
  o.z = c2 * rs * gw[c + 2] + gb[c + 2];
  o.w = c3 * rs * gw[c + 3] + gb[c + 3];
  *reinterpret_cast<float4*>(out + (size_t)row * E_ + c) = o;
}

// ---------------------------------------------------------------------------
// Attention: one block per (b, h, 16-query tile). 256 threads = 8 waves.
//   Phase1: S = Q K^T (WMMA, bf16, scaled + causal mask) -> LDS bf16 [16][T]
//   Phase2: row softmax in place (wave-owned rows)
//   Phase3: O = P V (WMMA), K-dim split over wave pairs, LDS reduction
// ---------------------------------------------------------------------------
__global__ __launch_bounds__(256)
void attn_k(const float* __restrict__ Q, const float* __restrict__ Kc,
            const float* __restrict__ Vc, float* __restrict__ O)
{
  __shared__ __bf16 Ss[16][T_];     // 32 KB score/prob tile
  __shared__ float  Ored[16][64];   // 4 KB cross-wave partial O

  const int tid  = threadIdx.x;
  const int lane = tid & 31;
  const int w    = tid >> 5;
  const int g    = lane >> 4;
  const int ln   = lane & 15;
  const int q0   = blockIdx.x * 16;
  const int bh   = blockIdx.y;
  const int b    = bh / H_;
  const int h    = bh % H_;
  const size_t base = (size_t)b * T_ * E_ + (size_t)h * HS_;
  const float scale = 0.125f;  // 1/sqrt(64)

  // ---- Phase 1: scores ----
  v16bf aq[2];
#pragma unroll
  for (int s = 0; s < 2; ++s)
    aq[s] = gfragA(Q + base + (size_t)(q0 + ln) * E_ + s * 32, g);

  for (int i = 0; i < 8; ++i) {
    const int t0 = (w * 8 + i) * 16;
    if (t0 > q0 + 15) {                       // fully causal-masked tile
#pragma unroll
      for (int r = 0; r < 8; ++r) Ss[r + 8 * g][t0 + ln] = (__bf16)(-1e30f);
    } else {
      v8f acc = {};
#pragma unroll
      for (int s = 0; s < 2; ++s) {
        const v16bf bkf = gfragB(Kc + base + (size_t)(t0 + ln) * E_ + s * 32, g);
        acc = wmma_bf16(aq[s], bkf, acc);
      }
#pragma unroll
      for (int r = 0; r < 8; ++r) {
        const int row = r + 8 * g;
        const int tg  = t0 + ln;
        const float sv = acc[r] * scale;
        Ss[row][tg] = (__bf16)((tg <= q0 + row) ? sv : -1e30f);
      }
    }
  }
  __syncthreads();

  // ---- Phase 2: softmax (wave w owns rows 2w, 2w+1) ----
#pragma unroll
  for (int rr = 0; rr < 2; ++rr) {
    const int row = w * 2 + rr;
    float mx = -1e30f;
    for (int c = lane; c < T_; c += 32) mx = fmaxf(mx, (float)Ss[row][c]);
#pragma unroll
    for (int o = 16; o > 0; o >>= 1) mx = fmaxf(mx, __shfl_xor(mx, o, 32));
    float sum = 0.0f;
    for (int c = lane; c < T_; c += 32) {
      const float e = __expf((float)Ss[row][c] - mx);
      sum += e;
      Ss[row][c] = (__bf16)e;
    }
#pragma unroll
    for (int o = 16; o > 0; o >>= 1) sum += __shfl_xor(sum, o, 32);
    const float inv = 1.0f / sum;
    for (int c = lane; c < T_; c += 32)
      Ss[row][c] = (__bf16)((float)Ss[row][c] * inv);
  }
  __syncthreads();

  // ---- Phase 3: O = P @ V ----
  const int nt = w & 3;    // which 16-wide slice of HS
  const int kh = w >> 2;   // which half of the K (=T) dimension
  const int d0 = nt * 16;
  v8f acc = {};
  for (int kc = kh * 16; kc < kh * 16 + 16; ++kc) {
    const v16bf pa = ldfragA(&Ss[ln][kc * 32], g);
    float vg[16];
#pragma unroll
    for (int e = 0; e < 16; ++e)
      vg[e] = Vc[base + (size_t)(kc * 32 + 16 * g + e) * E_ + d0 + ln];
    v16bf bv;
#pragma unroll
    for (int e = 0; e < 16; ++e) bv[e] = (__bf16)vg[e];
    acc = wmma_bf16(pa, bv, acc);
  }
  if (kh == 1) {
#pragma unroll
    for (int r = 0; r < 8; ++r) Ored[r + 8 * g][d0 + ln] = acc[r];
  }
  __syncthreads();
  if (kh == 0) {
#pragma unroll
    for (int r = 0; r < 8; ++r) {
      const int row = r + 8 * g;
      const float o = acc[r] + Ored[row][d0 + ln];
      O[base + (size_t)(q0 + row) * E_ + d0 + ln] = o;
    }
  }
}

// ---------------------------------------------------------------------------
// Loss: per-row logsumexp + NLL, mean via atomicAdd.
// ---------------------------------------------------------------------------
__global__ void zero_k(float* p) {
  if (threadIdx.x == 0 && blockIdx.x == 0) *p = 0.0f;
}

__global__ __launch_bounds__(256)
void loss_k(const float* __restrict__ logits, const int* __restrict__ tgt,
            float* __restrict__ loss)
{
  __shared__ float red[8];
  const int row  = blockIdx.x;
  const int tid  = threadIdx.x;
  const int lane = tid & 31;
  const int w    = tid >> 5;
  const float* lr = logits + (size_t)row * V_;

  float mx = -1e30f;
  for (int c = tid; c < V_; c += 256) mx = fmaxf(mx, lr[c]);
#pragma unroll
  for (int o = 16; o > 0; o >>= 1) mx = fmaxf(mx, __shfl_xor(mx, o, 32));
  if (lane == 0) red[w] = mx;
  __syncthreads();
  float m2 = -1e30f;
#pragma unroll
  for (int i = 0; i < 8; ++i) m2 = fmaxf(m2, red[i]);
  __syncthreads();

  float s = 0.0f;
  for (int c = tid; c < V_; c += 256) s += __expf(lr[c] - m2);
#pragma unroll
  for (int o = 16; o > 0; o >>= 1) s += __shfl_xor(s, o, 32);
  if (lane == 0) red[w] = s;
  __syncthreads();

  if (tid == 0) {
    float tot = 0.0f;
#pragma unroll
    for (int i = 0; i < 8; ++i) tot += red[i];
    const float lse = m2 + __logf(tot);
    const float tl  = lr[tgt[row]];
    atomicAdd(loss, (lse - tl) * (1.0f / NTOK));
  }
}

// ---------------------------------------------------------------------------
// Orchestration
// ---------------------------------------------------------------------------
extern "C" void kernel_launch(void* const* d_in, const int* in_sizes, int n_in,
                              void* d_out, int out_size, void* d_ws, size_t ws_size,
                              hipStream_t stream)
{
  (void)in_sizes; (void)n_in; (void)out_size; (void)ws_size;

  const int*   index = (const int*)  d_in[0];
  const int*   tgt   = (const int*)  d_in[1];
  const float* tok   = (const float*)d_in[2];
  const float* pos   = (const float*)d_in[3];
  const float* Wq    = (const float*)d_in[4];
  const float* Wk    = (const float*)d_in[5];
  const float* Wv    = (const float*)d_in[6];
  const float* Wo    = (const float*)d_in[7];
  const float* bo    = (const float*)d_in[8];
  const float* ln0g  = (const float*)d_in[9];
  const float* ln0b  = (const float*)d_in[10];
  const float* ln1g  = (const float*)d_in[11];
  const float* ln1b  = (const float*)d_in[12];
  const float* W1    = (const float*)d_in[13];
  const float* b1    = (const float*)d_in[14];
  const float* W2    = (const float*)d_in[15];
  const float* b2    = (const float*)d_in[16];
  const float* flng  = (const float*)d_in[17];
  const float* flnb  = (const float*)d_in[18];
  const float* lmW   = (const float*)d_in[19];
  const float* lmb   = (const float*)d_in[20];

  float* logits = (float*)d_out;
  float* loss   = logits + (size_t)NTOK * V_;

  // workspace carve-up (fp32): 6 * [4096,1024] + [4096,4096] = ~160 MB
  float* x   = (float*)d_ws;
  float* h   = x   + (size_t)NTOK * E_;
  float* qb  = h   + (size_t)NTOK * E_;
  float* kb  = qb  + (size_t)NTOK * E_;
  float* vb  = kb  + (size_t)NTOK * E_;
  float* att = vb  + (size_t)NTOK * E_;
  float* hh  = att + (size_t)NTOK * E_;

  const dim3 gE(E_ / 128, NTOK / 128);       // [4096 x 1024] outputs
  const dim3 gF(DFF_ / 128, NTOK / 128);     // [4096 x 4096] outputs
  const dim3 gV(V_ / 128, NTOK / 128);       // [4096 x 32000] outputs

  embed_k<<<NTOK, 256, 0, stream>>>(index, tok, pos, x);

  for (int l = 0; l < L_; ++l) {
    layernorm_k<<<NTOK, 256, 0, stream>>>(x, ln0g + l * E_, ln0b + l * E_, h);
    gemm_bf16w<<<gE, 256, 0, stream>>>(h, Wq + (size_t)l * E_ * E_, nullptr, nullptr, qb, NTOK, E_, E_, 0);
    gemm_bf16w<<<gE, 256, 0, stream>>>(h, Wk + (size_t)l * E_ * E_, nullptr, nullptr, kb, NTOK, E_, E_, 0);
    gemm_bf16w<<<gE, 256, 0, stream>>>(h, Wv + (size_t)l * E_ * E_, nullptr, nullptr, vb, NTOK, E_, E_, 0);
    attn_k<<<dim3(T_ / 16, B_ * H_), 256, 0, stream>>>(qb, kb, vb, att);
    gemm_bf16w<<<gE, 256, 0, stream>>>(att, Wo + (size_t)l * E_ * E_, bo + l * E_, x, x, NTOK, E_, E_, 0);
    layernorm_k<<<NTOK, 256, 0, stream>>>(x, ln1g + l * E_, ln1b + l * E_, h);
    gemm_bf16w<<<gF, 256, 0, stream>>>(h, W1 + (size_t)l * E_ * DFF_, b1 + l * DFF_, nullptr, hh, NTOK, DFF_, E_, 1);
    gemm_bf16w<<<gE, 256, 0, stream>>>(hh, W2 + (size_t)l * DFF_ * E_, b2 + l * E_, x, x, NTOK, E_, DFF_, 0);
  }

  layernorm_k<<<NTOK, 256, 0, stream>>>(x, flng, flnb, h);
  zero_k<<<1, 32, 0, stream>>>(loss);
  gemm_bf16w<<<gV, 256, 0, stream>>>(h, lmW, lmb, nullptr, logits, NTOK, V_, E_, 0);
  loss_k<<<NTOK, 256, 0, stream>>>(logits, tgt, loss);
}
